// LlamaPerformerAttention_11364483465686
// MI455X (gfx1250) — compile-verified, matrix operand order checked by reading, not run on hardware
//
#include <hip/hip_runtime.h>
#include <hip/hip_bf16.h>

// ---------------------------------------------------------------------------
// Problem constants (from the reference)
// ---------------------------------------------------------------------------
#define BB   2
#define SS   4096
#define HID  2048
#define HH   16
#define HKV  8
#define DD   128
#define MM   256
#define CHUNK 128
#define NCHUNK (SS / CHUNK)
#define EPSF 1e-6f
#define DQ_SCALE 0.29730177875068026f   /* D^-0.25 = 128^-0.25 */
#define RM_SCALE 0.0625f                /* M^-0.5  = 256^-0.5  */

typedef __attribute__((ext_vector_type(16))) __bf16 bf16x16;
typedef __attribute__((ext_vector_type(8)))  __bf16 bf16x8;
typedef __attribute__((ext_vector_type(8)))  float  v8f;

// ---------------------------------------------------------------------------
// WMMA helpers (CDNA5: D = A(16xK) * B(Kx16) + C, wave32, K=32 bf16)
//   A fragment: lane&15 = row,   K = (lane>>4)*8 + e (+8 for e>=8)  -> two
//               contiguous 8-element (16B) runs per lane.
//   B fragment: lane&15 = col N, same K pattern -> operand stored (N,K)
//               row-major loads as two 16B runs per lane.
//   C fragment: vgpr r -> row = r + (lane>>4)*8, col = lane&15.
// ---------------------------------------------------------------------------
__device__ __forceinline__ v8f wmma_bf16(bf16x16 a, bf16x16 b, v8f c) {
  return __builtin_amdgcn_wmma_f32_16x16x32_bf16(
      /*neg_a=*/false, a, /*neg_b=*/false, b,
      /*c_mod=*/(short)0, c, /*reuse_a=*/false, /*reuse_b=*/false);
}

// Load a 16xK(=32) fragment from a row-major (row, K-contiguous) bf16 source.
// `base` points at (row0, k0); `ld` is the leading dimension in elements
// (multiple of 8 so the 16B loads stay aligned).
__device__ __forceinline__ bf16x16 load_frag(const __bf16* base, size_t ld, int lane) {
  const __bf16* p = base + (size_t)(lane & 15) * ld + (size_t)((lane >> 4) << 3);
  bf16x8 lo = *reinterpret_cast<const bf16x8*>(p);
  bf16x8 hi = *reinterpret_cast<const bf16x8*>(p + 16);
  bf16x16 r;
#pragma unroll
  for (int i = 0; i < 8; ++i) { r[i] = lo[i]; r[i + 8] = hi[i]; }
  return r;
}

// ---------------------------------------------------------------------------
// fp32 -> bf16 staging
// ---------------------------------------------------------------------------
__global__ void cvt_f32_to_bf16(const float* __restrict__ s,
                                __bf16* __restrict__ d, long long n) {
  long long i = (long long)blockIdx.x * blockDim.x + threadIdx.x;
  long long stride = (long long)gridDim.x * blockDim.x;
  for (; i < n; i += stride) d[i] = (__bf16)s[i];
}

__global__ void init_gmax(unsigned int* g) { *g = 0u; }

// Order-preserving float->uint map so atomicMax works for signed floats.
__device__ __forceinline__ unsigned int enc_f32(float v) {
  unsigned int b = __float_as_uint(v);
  return (b & 0x80000000u) ? ~b : (b | 0x80000000u);
}
__device__ __forceinline__ float dec_f32(unsigned int b) {
  b = (b & 0x80000000u) ? (b & 0x7FFFFFFFu) : ~b;
  return __uint_as_float(b);
}

// ---------------------------------------------------------------------------
// Generic bf16 GEMM:  C[Mr,N] = A[Mr,K] * Bw[N,K]^T   (Bw row-major (N,K))
// Block tile 128x128, 8 waves (4 row-bands x 2 col-bands), each wave 32x64.
// ---------------------------------------------------------------------------
template <typename OutT>
__global__ __launch_bounds__(256) void gemm_bf16_nt(
    const __bf16* __restrict__ A, const __bf16* __restrict__ Bw,
    OutT* __restrict__ C, int Mr, int N, int K) {
  const int lane = threadIdx.x & 31;
  const int wave = threadIdx.x >> 5;
  const int wm = wave & 3;      // 0..3 -> 32-row band
  const int wn = wave >> 2;     // 0..1 -> 64-col band
  const int row0 = blockIdx.x * 128 + wm * 32;
  const int col0 = blockIdx.y * 128 + wn * 64;

  v8f acc[2][4];
#pragma unroll
  for (int i = 0; i < 2; ++i)
#pragma unroll
    for (int t = 0; t < 4; ++t) acc[i][t] = {};

  for (int ks = 0; ks < K; ks += 32) {
    if (ks + 64 < K) {  // speculative L2 prefetch of the next A panel
      __builtin_prefetch(A + (size_t)row0 * K + ks + 64, 0, 1);
    }
    bf16x16 a0 = load_frag(A + (size_t)row0 * K + ks, (size_t)K, lane);
    bf16x16 a1 = load_frag(A + (size_t)(row0 + 16) * K + ks, (size_t)K, lane);
#pragma unroll
    for (int t = 0; t < 4; ++t) {
      bf16x16 bf = load_frag(Bw + (size_t)(col0 + t * 16) * K + ks, (size_t)K, lane);
      acc[0][t] = wmma_bf16(a0, bf, acc[0][t]);
      acc[1][t] = wmma_bf16(a1, bf, acc[1][t]);
    }
  }

#pragma unroll
  for (int i = 0; i < 2; ++i)
#pragma unroll
    for (int t = 0; t < 4; ++t)
#pragma unroll
      for (int r = 0; r < 8; ++r) {
        int row = row0 + i * 16 + r + ((lane >> 4) << 3);
        int col = col0 + t * 16 + (lane & 15);
        C[(size_t)row * N + col] = (OutT)acc[i][t][r];
      }
}

// ---------------------------------------------------------------------------
// RoPE + D^-0.25 scaling + 0.5*||xn||^2 row sums.  One block per (b,s,head).
// ---------------------------------------------------------------------------
__global__ __launch_bounds__(128) void rope_scale_kernel(
    const __bf16* __restrict__ Q, const __bf16* __restrict__ Kb,
    const float* __restrict__ COS, const float* __restrict__ SIN,
    __bf16* __restrict__ XQ, __bf16* __restrict__ XK,
    float* __restrict__ SQQ, float* __restrict__ SQK) {
  const int row = blockIdx.x;                // B*S*(H+HKV)
  const int d = threadIdx.x;                 // 0..127
  const int bs = row / (HH + HKV);
  const int r = row % (HH + HKV);
  const float c = COS[(size_t)bs * DD + d];
  const float sn = SIN[(size_t)bs * DD + d];

  float x, xr;
  if (r < HH) {
    const __bf16* q = Q + (size_t)bs * HH * DD + (size_t)r * DD;
    x = (float)q[d];
    xr = (d < DD / 2) ? -(float)q[d + DD / 2] : (float)q[d - DD / 2];
  } else {
    const int hk = r - HH;
    const __bf16* k = Kb + (size_t)bs * HKV * DD + (size_t)hk * DD;
    x = (float)k[d];
    xr = (d < DD / 2) ? -(float)k[d + DD / 2] : (float)k[d - DD / 2];
  }
  const float xn = (x * c + xr * sn) * DQ_SCALE;

  float ss = xn * xn;
#pragma unroll
  for (int off = 16; off; off >>= 1) ss += __shfl_xor(ss, off);
  __shared__ float sred[4];
  if ((threadIdx.x & 31) == 0) sred[threadIdx.x >> 5] = ss;
  __syncthreads();
  const float tot = sred[0] + sred[1] + sred[2] + sred[3];

  if (r < HH) {
    XQ[((size_t)bs * HH + r) * DD + d] = (__bf16)xn;
    if (threadIdx.x == 0) SQQ[(size_t)bs * HH + r] = 0.5f * tot;
  } else {
    const int hk = r - HH;
    XK[((size_t)bs * HKV + hk) * DD + d] = (__bf16)xn;
    if (threadIdx.x == 0) SQK[(size_t)bs * HKV + hk] = 0.5f * tot;
  }
}

// ---------------------------------------------------------------------------
// Feature-map finalization.
// ---------------------------------------------------------------------------
__global__ __launch_bounds__(256) void phi_q_finalize(
    const __bf16* __restrict__ U, const float* __restrict__ SQ,
    __bf16* __restrict__ PQ) {
  const int row = blockIdx.x;                // B*S*H
  const int m = threadIdx.x;                 // 0..255 == M
  const float z = (float)U[(size_t)row * MM + m] - SQ[row];
  float mx = z;
#pragma unroll
  for (int off = 16; off; off >>= 1) mx = fmaxf(mx, __shfl_xor(mx, off));
  __shared__ float smax[8];
  if ((threadIdx.x & 31) == 0) smax[threadIdx.x >> 5] = mx;
  __syncthreads();
  float rm = smax[0];
#pragma unroll
  for (int i = 1; i < 8; ++i) rm = fmaxf(rm, smax[i]);
  PQ[(size_t)row * MM + m] = (__bf16)(__expf(z - rm) * RM_SCALE + EPSF);
}

__global__ __launch_bounds__(256) void phi_k_maxpass(
    const __bf16* __restrict__ U, const float* __restrict__ SQ,
    unsigned int* __restrict__ gmax) {
  const int row = blockIdx.x;                // B*S*HKV
  const int m = threadIdx.x;
  const float z = (float)U[(size_t)row * MM + m] - SQ[row];
  float mx = z;
#pragma unroll
  for (int off = 16; off; off >>= 1) mx = fmaxf(mx, __shfl_xor(mx, off));
  __shared__ float smax[8];
  if ((threadIdx.x & 31) == 0) smax[threadIdx.x >> 5] = mx;
  __syncthreads();
  if (threadIdx.x == 0) {
    float bm = smax[0];
#pragma unroll
    for (int i = 1; i < 8; ++i) bm = fmaxf(bm, smax[i]);
    atomicMax(gmax, enc_f32(bm));
  }
}

__global__ __launch_bounds__(256) void phi_k_finalize(
    const __bf16* __restrict__ U, const float* __restrict__ SQ,
    const unsigned int* __restrict__ gmax, __bf16* __restrict__ PK) {
  const int row = blockIdx.x;
  const int m = threadIdx.x;
  const float gm = dec_f32(*gmax);
  const float z = (float)U[(size_t)row * MM + m] - SQ[row];
  PK[(size_t)row * MM + m] = (__bf16)(__expf(z - gm) * RM_SCALE + EPSF);
}

// ---------------------------------------------------------------------------
// Parallel linear-attention scan, pass A:  per (b,h,chunk) compute the local
// rank-128 update  KV_c[d,m] = sum_c V[c,d]*phi_k[c,m]  (WMMA, D x M tile) and
// ksum_c[m] = column sums of phi_k.  B*H*NCHUNK = 1024 independent blocks.
// ---------------------------------------------------------------------------
__global__ __launch_bounds__(256) void chunk_kv_kernel(
    const __bf16* __restrict__ PK, const __bf16* __restrict__ V,
    __bf16* __restrict__ KV, float* __restrict__ KSUM) {
  __shared__ __bf16 ldsVT[CHUNK * CHUNK];   // V^T chunk (d-major)
  __shared__ __bf16 ldsKT[CHUNK * CHUNK];   // phi_k^T half (m-major)

  const int bc = blockIdx.x;                // bh*NCHUNK + chunk
  const int bh = bc / NCHUNK;
  const int chunk = bc % NCHUNK;
  const int b = bh / HH, h = bh % HH, hk = h >> 1;
  const int tid = threadIdx.x, lane = tid & 31, wave = tid >> 5;
  const int s0 = chunk * CHUNK;
  const size_t pk_ld = (size_t)HKV * MM;    // 2048
  const size_t v_ld = (size_t)HKV * DD;     // 1024
  const __bf16* pk0 = PK + ((size_t)(b * SS + s0) * HKV + hk) * MM;
  const __bf16* v0 = V + (size_t)(b * SS + s0) * v_ld + (size_t)hk * DD;
  __bf16* kv = KV + (size_t)bc * DD * MM;   // (D rows, M cols) row-major
  float* ks = KSUM + (size_t)bc * MM;

  for (int i = tid; i < CHUNK * DD; i += 256) {
    const int c = i >> 7, d = i & 127;
    ldsVT[d * CHUNK + c] = v0[(size_t)c * v_ld + d];
  }
  for (int half = 0; half < 2; ++half) {
    __syncthreads();                        // vT ready / prior ldsKT reads done
    for (int i = tid; i < CHUNK * 128; i += 256) {
      const int c = i >> 7, mloc = i & 127; // coalesced reads of pk rows
      ldsKT[mloc * CHUNK + c] = pk0[(size_t)c * pk_ld + half * 128 + mloc];
    }
    __syncthreads();
    if (tid < 128) {                        // ksum from LDS, contiguous over c
      float acc = 0.f;
      for (int c = 0; c < CHUNK; ++c) acc += (float)ldsKT[tid * CHUNK + c];
      ks[half * 128 + tid] = acc;
    }
#pragma unroll
    for (int t = 0; t < 8; ++t) {           // wave owns d rows 16w..16w+15
      v8f acc = {};
      for (int kk = 0; kk < CHUNK; kk += 32) {
        bf16x16 af = load_frag(ldsVT + (size_t)(wave * 16) * CHUNK + kk, (size_t)CHUNK, lane);
        bf16x16 bfrag = load_frag(ldsKT + (size_t)(t * 16) * CHUNK + kk, (size_t)CHUNK, lane);
        acc = wmma_bf16(af, bfrag, acc);
      }
#pragma unroll
      for (int r = 0; r < 8; ++r) {
        const int d = wave * 16 + r + ((lane >> 4) << 3);
        const int m = half * 128 + t * 16 + (lane & 15);
        kv[(size_t)d * MM + m] = (__bf16)acc[r];
      }
    }
  }
}

// ---------------------------------------------------------------------------
// Pass B: in-place exclusive prefix scan over the 32 chunks.  fp32 carry,
// bf16 storage.  Pure streaming: ~134 MB at 23.3 TB/s ≈ 6 us.
// ---------------------------------------------------------------------------
__global__ __launch_bounds__(256) void kv_prefix_scan(__bf16* __restrict__ KV) {
  const size_t idx = (size_t)blockIdx.x * blockDim.x + threadIdx.x; // B*H*D*M
  const size_t per = (size_t)DD * MM;
  const size_t bh = idx / per;
  const size_t e = idx % per;
  __bf16* base = KV + bh * NCHUNK * per + e;
  float carry = 0.f;
  for (int c = 0; c < NCHUNK; ++c) {
    const float t = (float)base[(size_t)c * per];
    base[(size_t)c * per] = (__bf16)carry;
    carry += t;
  }
}

__global__ __launch_bounds__(256) void ksum_prefix_scan(float* __restrict__ KS) {
  const int idx = blockIdx.x * blockDim.x + threadIdx.x;            // B*H*M
  const int bh = idx / MM, m = idx % MM;
  float* base = KS + (size_t)bh * NCHUNK * MM + m;
  float carry = 0.f;
  for (int c = 0; c < NCHUNK; ++c) {
    const float t = base[(size_t)c * MM];
    base[(size_t)c * MM] = carry;
    carry += t;
  }
}

// ---------------------------------------------------------------------------
// Pass C: per (b,h,chunk) produce outputs.
//   A   = phi_q * phi_k^T   (causal-masked, 128x128, K=M)
//   num = phi_q * S_prev  +  A_masked * V
//   den = phi_q . ksum_prev + rowsum(A_masked) + eps
// 1024 independent blocks, 8 waves, each wave owns a 16-row band.
// ---------------------------------------------------------------------------
__global__ __launch_bounds__(256) void chunk_out_kernel(
    const __bf16* __restrict__ PQ, const __bf16* __restrict__ PK,
    const __bf16* __restrict__ V, const __bf16* __restrict__ KV,
    const float* __restrict__ KSUM, __bf16* __restrict__ OUT) {
  __shared__ __bf16 ldsA[CHUNK * CHUNK];    // masked QK^T tile
  __shared__ __bf16 ldsVT[CHUNK * CHUNK];   // transposed V chunk

  const int bc = blockIdx.x;                // bh*NCHUNK + chunk
  const int bh = bc / NCHUNK;
  const int chunk = bc % NCHUNK;
  const int b = bh / HH, h = bh % HH, hk = h >> 1;
  const int tid = threadIdx.x, lane = tid & 31, wave = tid >> 5;
  const int s0 = chunk * CHUNK;
  const size_t pq_ld = (size_t)HH * MM;     // 4096
  const size_t pk_ld = (size_t)HKV * MM;    // 2048
  const size_t v_ld = (size_t)HKV * DD;     // 1024
  const __bf16* pq0 = PQ + ((size_t)(b * SS + s0) * HH + h) * MM;
  const __bf16* pk0 = PK + ((size_t)(b * SS + s0) * HKV + hk) * MM;
  const __bf16* v0 = V + (size_t)(b * SS + s0) * v_ld + (size_t)hk * DD;
  const __bf16* st = KV + (size_t)bc * DD * MM;   // exclusive-prefix state
  const float* kpre = KSUM + (size_t)bc * MM;     // exclusive-prefix ksum

  // stage V^T
  for (int i = tid; i < CHUNK * DD; i += 256) {
    const int c = i >> 7, d = i & 127;
    ldsVT[d * CHUNK + c] = v0[(size_t)c * v_ld + d];
  }
  __syncthreads();

  // ---- A = phi_q * phi_k^T  (K = M = 256)
  v8f aq[8];
#pragma unroll
  for (int t = 0; t < 8; ++t) aq[t] = {};
  for (int ks = 0; ks < MM; ks += 32) {
    bf16x16 af = load_frag(pq0 + (size_t)(wave * 16) * pq_ld + ks, pq_ld, lane);
#pragma unroll
    for (int t = 0; t < 8; ++t) {
      bf16x16 bfrag = load_frag(pk0 + (size_t)(t * 16) * pk_ld + ks, pk_ld, lane);
      aq[t] = wmma_bf16(af, bfrag, aq[t]);
    }
  }
  // causal mask (inclusive) -> bf16 tile in LDS
#pragma unroll
  for (int t = 0; t < 8; ++t)
#pragma unroll
    for (int r = 0; r < 8; ++r) {
      const int c = wave * 16 + r + ((lane >> 4) << 3);
      const int c2 = t * 16 + (lane & 15);
      ldsA[c * CHUNK + c2] = (__bf16)((c2 <= c) ? aq[t][r] : 0.f);
    }
  __syncthreads();

  // ---- numerator = phi_q * S_prev  +  A_masked * V
  v8f an[8];
#pragma unroll
  for (int t = 0; t < 8; ++t) an[t] = {};
  for (int ks = 0; ks < MM; ks += 32) {             // inter-chunk (K = M)
    bf16x16 af = load_frag(pq0 + (size_t)(wave * 16) * pq_ld + ks, pq_ld, lane);
#pragma unroll
    for (int t = 0; t < 8; ++t) {
      bf16x16 bfrag = load_frag(st + (size_t)(t * 16) * MM + ks, (size_t)MM, lane);
      an[t] = wmma_bf16(af, bfrag, an[t]);
    }
  }
  for (int ks = 0; ks < CHUNK; ks += 32) {          // intra-chunk (K = CHUNK)
    bf16x16 af = load_frag(ldsA + (size_t)(wave * 16) * CHUNK + ks, (size_t)CHUNK, lane);
#pragma unroll
    for (int t = 0; t < 8; ++t) {
      bf16x16 bfrag = load_frag(ldsVT + (size_t)(t * 16) * CHUNK + ks, (size_t)CHUNK, lane);
      an[t] = wmma_bf16(af, bfrag, an[t]);
    }
  }

  // ---- denominator for this wave's 16-row band (dup across half-waves)
  float dv = EPSF;
  {
    const int c = wave * 16 + (lane & 15);
    const __bf16* pqr = pq0 + (size_t)c * pq_ld;
    for (int m = 0; m < MM; ++m) dv += (float)pqr[m] * kpre[m];
    for (int c2 = 0; c2 <= c; ++c2) dv += (float)ldsA[c * CHUNK + c2];
  }

  // ---- out[c,d] = num / den, bf16, layout (b, s, h, d)
#pragma unroll
  for (int t = 0; t < 8; ++t)
#pragma unroll
    for (int r = 0; r < 8; ++r) {
      const int rb = r + ((lane >> 4) << 3);        // 0..15 within band
      const int c = wave * 16 + rb;
      const int d = t * 16 + (lane & 15);
      const float den = __shfl(dv, rb);             // lane rb holds den(row rb)
      OUT[((size_t)(b * SS + s0 + c) * HH + h) * DD + d] = (__bf16)(an[t][r] / den);
    }
}

// ---------------------------------------------------------------------------
// Host-side orchestration
// ---------------------------------------------------------------------------
static inline void launch_cvt(const float* s, __bf16* d, long long n, hipStream_t st) {
  int blocks = (int)((n + 255) / 256);
  if (blocks > 8192) blocks = 8192;
  cvt_f32_to_bf16<<<blocks, 256, 0, st>>>(s, d, n);
}

extern "C" void kernel_launch(void* const* d_in, const int* in_sizes, int n_in,
                              void* d_out, int out_size, void* d_ws, size_t ws_size,
                              hipStream_t stream) {
  (void)in_sizes; (void)n_in; (void)out_size; (void)ws_size;
  const float* hs   = (const float*)d_in[0];   // (B,S,HID)
  const float* cosb = (const float*)d_in[1];   // (B,S,D)
  const float* sinb = (const float*)d_in[2];   // (B,S,D)
  const float* wq   = (const float*)d_in[3];   // (H*D, HID)
  const float* wk   = (const float*)d_in[4];   // (HKV*D, HID)
  const float* wv   = (const float*)d_in[5];   // (HKV*D, HID)
  const float* wo   = (const float*)d_in[6];   // (HID, H*D)
  const float* pm   = (const float*)d_in[7];   // (M, D)
  float* out = (float*)d_out;                  // (B,S,HID)

  // --- workspace carve-up (bf16 staging + intermediates), 256B aligned
  char* ws = (char*)d_ws;
  size_t cur = 0;
  auto alloc = [&](size_t bytes) -> char* {
    char* p = ws + cur;
    cur = (cur + bytes + 255) & ~(size_t)255;
    return p;
  };
  const size_t R = (size_t)BB * SS;            // 8192 token rows
  __bf16* hb    = (__bf16*)alloc(R * HID * 2);
  __bf16* wqb   = (__bf16*)alloc((size_t)HH * DD * HID * 2);
  __bf16* wkb   = (__bf16*)alloc((size_t)HKV * DD * HID * 2);
  __bf16* wvb   = (__bf16*)alloc((size_t)HKV * DD * HID * 2);
  __bf16* wob   = (__bf16*)alloc((size_t)HID * HH * DD * 2);
  __bf16* pmb   = (__bf16*)alloc((size_t)MM * DD * 2);
  __bf16* qbuf  = (__bf16*)alloc(R * HH * DD * 2);
  __bf16* kbuf  = (__bf16*)alloc(R * HKV * DD * 2);
  __bf16* vbuf  = (__bf16*)alloc(R * HKV * DD * 2);
  __bf16* xq    = (__bf16*)alloc(R * HH * DD * 2);
  __bf16* xk    = (__bf16*)alloc(R * HKV * DD * 2);
  float*  sqq   = (float*) alloc(R * HH * 4);
  float*  sqk   = (float*) alloc(R * HKV * 4);
  __bf16* uq    = (__bf16*)alloc(R * HH * MM * 2);
  __bf16* uk    = (__bf16*)alloc(R * HKV * MM * 2);
  __bf16* pq    = (__bf16*)alloc(R * HH * MM * 2);
  __bf16* pk    = (__bf16*)alloc(R * HKV * MM * 2);
  __bf16* attn  = (__bf16*)alloc(R * HH * DD * 2);
  __bf16* kvbuf = (__bf16*)alloc((size_t)BB * HH * NCHUNK * DD * MM * 2); // 67MB
  float*  ksbuf = (float*) alloc((size_t)BB * HH * NCHUNK * MM * 4);      // 1MB
  unsigned int* gmax = (unsigned int*)alloc(256);

  // 1) fp32 -> bf16 staging
  launch_cvt(hs, hb, (long long)R * HID, stream);
  launch_cvt(wq, wqb, (long long)HH * DD * HID, stream);
  launch_cvt(wk, wkb, (long long)HKV * DD * HID, stream);
  launch_cvt(wv, wvb, (long long)HKV * DD * HID, stream);
  launch_cvt(wo, wob, (long long)HID * HH * DD, stream);
  launch_cvt(pm, pmb, (long long)MM * DD, stream);
  init_gmax<<<1, 1, 0, stream>>>(gmax);

  // 2) QKV projections  (C = hidden * W^T, weights already (N,K) row-major)
  gemm_bf16_nt<__bf16><<<dim3(R / 128, (HH * DD) / 128), 256, 0, stream>>>(
      hb, wqb, qbuf, (int)R, HH * DD, HID);
  gemm_bf16_nt<__bf16><<<dim3(R / 128, (HKV * DD) / 128), 256, 0, stream>>>(
      hb, wkb, kbuf, (int)R, HKV * DD, HID);
  gemm_bf16_nt<__bf16><<<dim3(R / 128, (HKV * DD) / 128), 256, 0, stream>>>(
      hb, wvb, vbuf, (int)R, HKV * DD, HID);

  // 3) RoPE + D^-0.25 scaling + row sq-sums
  rope_scale_kernel<<<(unsigned)(R * (HH + HKV)), 128, 0, stream>>>(
      qbuf, kbuf, cosb, sinb, xq, xk, sqq, sqk);

  // 4) feature-map projection  U = xn * P^T   (P is (M,D) row-major = (N,K))
  gemm_bf16_nt<__bf16><<<dim3((R * HH) / 128, MM / 128), 256, 0, stream>>>(
      xq, pmb, uq, (int)(R * HH), MM, DD);
  gemm_bf16_nt<__bf16><<<dim3((R * HKV) / 128, MM / 128), 256, 0, stream>>>(
      xk, pmb, uk, (int)(R * HKV), MM, DD);

  // 5) phi finalization (q: per-row max; k: global max, two passes)
  phi_q_finalize<<<(unsigned)(R * HH), MM, 0, stream>>>(uq, sqq, pq);
  phi_k_maxpass<<<(unsigned)(R * HKV), MM, 0, stream>>>(uk, sqk, gmax);
  phi_k_finalize<<<(unsigned)(R * HKV), MM, 0, stream>>>(uk, sqk, gmax, pk);

  // 6) parallel chunked scan: local KV -> exclusive prefix -> outputs
  chunk_kv_kernel<<<BB * HH * NCHUNK, 256, 0, stream>>>(pk, vbuf, kvbuf, ksbuf);
  kv_prefix_scan<<<(unsigned)((BB * HH * (size_t)DD * MM) / 256), 256, 0, stream>>>(kvbuf);
  ksum_prefix_scan<<<(BB * HH * MM) / 256, 256, 0, stream>>>(ksbuf);
  chunk_out_kernel<<<BB * HH * NCHUNK, 256, 0, stream>>>(
      pq, pk, vbuf, kvbuf, ksbuf, attn);

  // 7) output projection: out = attn * wo^T  (fp32 result)
  gemm_bf16_nt<float><<<dim3(R / 128, HID / 128), 256, 0, stream>>>(
      attn, wob, out, (int)R, HID, HH * DD);
}
